// BiLevelGAT_37546604102311
// MI455X (gfx1250) — matrix operand order repaired and checked by reading, not compiled, not run on hardware
//
#include <hip/hip_runtime.h>
#include <math.h>

// ---------------------------------------------------------------------------
// BiLevelGAT for MI455X (gfx1250, wave32).
// Dense projections -> v_wmma_f32_16x16x32_bf16. Weights are pre-packed once
// per call into bf16 WMMA B-fragment order (per-lane contiguous 32B), so the
// GEMM hot loop is: 4x b128 A loads + packed f32->bf16 cvt of A + 2x b128 B
// loads per column tile + WMMA. One wave owns a 16-row stripe x all N columns
// with fused LN+GELU / attention-logit epilogues (ds_swizzle butterflies).
// Edge softmax/aggregation -> streaming kernels with f32 atomics (max/add).
// Workspace requirement: ~57.6M floats + 190KB packed weights (~230.6 MB).
// NOTE: all GEMM row counts (NU=100000, NI=50000) are multiples of 16 -> no
// per-row bounds checks anywhere in the stripe kernel.
// ---------------------------------------------------------------------------

typedef __attribute__((ext_vector_type(16))) __bf16 v16bf;
typedef __attribute__((ext_vector_type(2)))  __bf16 v2bf;
typedef __attribute__((ext_vector_type(8)))  float  v8f;

#define NU_   100000
#define NI_   50000
#define ES_N  1000000
#define EI_N  2000000

// ----- helpers -------------------------------------------------------------

__device__ __forceinline__ __bf16 f2bf(float f) {
  unsigned u = __builtin_bit_cast(unsigned, f);
  unsigned r = u + 0x7FFFu + ((u >> 16) & 1u);   // round-to-nearest-even
  unsigned short h = (unsigned short)(r >> 16);
  return __builtin_bit_cast(__bf16, h);
}

// packed f32x2 -> bf16x2 (single v_cvt_pk_bf16_f32 when available)
__device__ __forceinline__ void put2(v16bf& f, int idx, float lo, float hi) {
#if __has_builtin(__builtin_amdgcn_cvt_pk_bf16_f32)
  v2bf p = __builtin_amdgcn_cvt_pk_bf16_f32(lo, hi);
  f[idx] = p[0]; f[idx + 1] = p[1];
#else
  f[idx] = f2bf(lo); f[idx + 1] = f2bf(hi);
#endif
}

// butterfly add across the 16 lanes of one wave-half via ds_swizzle (xor mode:
// offset = {0, xor[14:10], or[9:5]=0, and[4:0]=0x1f}); masks 1/2/4/8 keep bit4.
template<int MASK>
__device__ __forceinline__ float swz_add(float v) {
  int x = __builtin_amdgcn_ds_swizzle(__builtin_bit_cast(int, v), (MASK << 10) | 0x1f);
  return v + __builtin_bit_cast(float, x);
}
__device__ __forceinline__ float half_reduce(float v) {
  v = swz_add<1>(v); v = swz_add<2>(v); v = swz_add<4>(v); v = swz_add<8>(v);
  return v;
}

__device__ __forceinline__ float eluf(float x)  { return x > 0.f ? x : expm1f(x); }
__device__ __forceinline__ float geluf(float x) { return 0.5f * x * (1.f + erff(x * 0.70710678118654752f)); }

// ----- weight pre-pack: W[K,N] f32 -> bf16 WMMA B fragments ----------------
// element j of the fragment for (kt, t, lane): K index kt*32+(lane>>4)*8+(j&7)+(j>>3)*16,
// column t*16+(lane&15); zero-padded to KT=ceil(K/32)*32.
// flat bf16 index = ((kt*NT + t)*32 + lane)*16 + j  (per-lane 32B contiguous)
template<int NT, int K>
__global__ void pack_w_k(const float* __restrict__ W, __bf16* __restrict__ P) {
  constexpr int N  = NT * 16;
  constexpr int KT = (K + 31) & ~31;
  constexpr int TOTAL = (KT / 32) * NT * 512;
  int idx = blockIdx.x * 256 + threadIdx.x;
  if (idx >= TOTAL) return;
  int j    = idx & 15;
  int lane = (idx >> 4) & 31;
  int t    = (idx >> 9) % NT;
  int kt   = idx / (512 * NT);
  int k    = kt * 32 + (lane >> 4) * 8 + (j & 7) + (j >> 3) * 16;
  int col  = t * 16 + (lane & 15);
  float f  = (k < K) ? W[(long)k * N + col] : 0.f;
  P[idx] = f2bf(f);
}

// ----- fused WMMA GEMM: one wave computes rows [m0, m0+16) x all N cols ----
// MODE 0: Y = X@W (+bias)            (bias may be null)
// MODE 1: Y = gelu(LN(X@W + bias))   (lng/lnb = LN gamma/beta)
// MODE 2: Y = X@W  and attention logits es/ed = (Y.reshape(H,C) * a).sum(C)
// Requires M % 16 == 0. Wp = packed bf16 weights (see pack_w_k).
template<int NT, int H, int MODE, int K>
__global__ __launch_bounds__(256) void gemm_k(
    const float* __restrict__ X, const __bf16* __restrict__ Wp_,
    const float* __restrict__ bias,
    const float* __restrict__ lng, const float* __restrict__ lnb,
    const float* __restrict__ av_s, const float* __restrict__ av_d,
    float* __restrict__ Y, float* __restrict__ es, float* __restrict__ ed,
    int M)
{
  constexpr int N  = NT * 16;
  constexpr int KT = (K + 31) & ~31;
  const int lane  = threadIdx.x & 31;
  const int wid   = threadIdx.x >> 5;
  const int m0    = (blockIdx.x * 8 + wid) * 16;
  if (m0 >= M) return;

  const v16bf* __restrict__ Wp = (const v16bf*)Wp_;
  const int colL  = lane & 15;
  const int halfU = lane >> 4;

  v8f acc[NT];
  #pragma unroll
  for (int t = 0; t < NT; ++t)
    #pragma unroll
    for (int i = 0; i < 8; ++i) acc[t][i] = 0.f;

  // A fragment: lanes 0-15 row m0+lane, K elems {0..7,16..23}; lanes 16-31 {8..15,24..31}
  const int  arow  = m0 + colL;                  // always < M (M % 16 == 0)
  const int  kaOff = halfU * 8;
  const float* __restrict__ Xr = X + (long)arow * K;

  for (int kt = 0; kt < KT / 32; ++kt) {
    const int k0 = kt * 32;
    v16bf afrag;
    if constexpr ((K & 31) == 0) {
      // (arow*K + k0 + kaOff) is a multiple of 8 floats -> 16B-aligned b128 loads
      const float4 a0 = *(const float4*)(Xr + k0 + kaOff);
      const float4 a1 = *(const float4*)(Xr + k0 + kaOff + 4);
      const float4 a2 = *(const float4*)(Xr + k0 + 16 + kaOff);
      const float4 a3 = *(const float4*)(Xr + k0 + 16 + kaOff + 4);
      put2(afrag, 0,  a0.x, a0.y); put2(afrag, 2,  a0.z, a0.w);
      put2(afrag, 4,  a1.x, a1.y); put2(afrag, 6,  a1.z, a1.w);
      put2(afrag, 8,  a2.x, a2.y); put2(afrag, 10, a2.z, a2.w);
      put2(afrag, 12, a3.x, a3.y); put2(afrag, 14, a3.z, a3.w);
    } else {
      // odd K: clamped address (always valid) + value select, no exec diamonds
      #pragma unroll
      for (int j = 0; j < 8; ++j) {
        int k1 = k0 + kaOff + j;
        int k2 = k0 + 16 + kaOff + j;
        float f1 = Xr[k1 < K ? k1 : 0]; f1 = (k1 < K) ? f1 : 0.f;
        float f2 = Xr[k2 < K ? k2 : 0]; f2 = (k2 < K) ? f2 : 0.f;
        afrag[j]     = f2bf(f1);
        afrag[j + 8] = f2bf(f2);
      }
    }
    #pragma unroll
    for (int t = 0; t < NT; ++t) {
      v16bf bfrag = Wp[(kt * NT + t) * 32 + lane];   // 32B aligned vector load
      acc[t] = __builtin_amdgcn_wmma_f32_16x16x32_bf16(
          false, afrag, false, bfrag, (short)0, acc[t], false, false);
    }
  }

  // C/D layout: element i of acc[t] -> row m0 + i + 8*halfU, col t*16 + colL
  const int rbase = m0 + halfU * 8;

  if constexpr (MODE == 0) {
    #pragma unroll
    for (int t = 0; t < NT; ++t) {
      float bv = bias ? bias[t * 16 + colL] : 0.f;
      #pragma unroll
      for (int i = 0; i < 8; ++i)
        Y[(long)(rbase + i) * N + t * 16 + colL] = acc[t][i] + bv;
    }
  } else if constexpr (MODE == 1) {
    #pragma unroll
    for (int t = 0; t < NT; ++t) {
      float bv = bias[t * 16 + colL];
      #pragma unroll
      for (int i = 0; i < 8; ++i) acc[t][i] += bv;
    }
    const float invN = 1.f / (float)N;
    #pragma unroll
    for (int i = 0; i < 8; ++i) {
      float s = 0.f, s2 = 0.f;
      #pragma unroll
      for (int t = 0; t < NT; ++t) { float v = acc[t][i]; s += v; s2 += v * v; }
      s  = half_reduce(s);
      s2 = half_reduce(s2);
      float mean = s * invN;
      float var  = s2 * invN - mean * mean;
      float rstd = rsqrtf(var + 1e-5f);
      #pragma unroll
      for (int t = 0; t < NT; ++t) {
        int col = t * 16 + colL;
        float v = (acc[t][i] - mean) * rstd * lng[col] + lnb[col];
        Y[(long)(rbase + i) * N + col] = geluf(v);
      }
    }
  } else { // MODE == 2: attention
    #pragma unroll
    for (int t = 0; t < NT; ++t)
      #pragma unroll
      for (int i = 0; i < 8; ++i)
        Y[(long)(rbase + i) * N + t * 16 + colL] = acc[t][i];
    constexpr int TPH = NT / H;     // column tiles per head
    constexpr int C   = TPH * 16;   // channels per head
    #pragma unroll
    for (int h = 0; h < H; ++h) {
      #pragma unroll
      for (int i = 0; i < 8; ++i) {
        float s = 0.f, sd = 0.f;
        #pragma unroll
        for (int tt = 0; tt < TPH; ++tt) {
          float v  = acc[h * TPH + tt][i];
          int   cc = tt * 16 + colL;
          if (av_s) s  += v * av_s[h * C + cc];
          if (av_d) sd += v * av_d[h * C + cc];
        }
        s  = half_reduce(s);
        sd = half_reduce(sd);
        if (colL == 0) {
          if (es) es[(long)(rbase + i) * H + h] = s;
          if (ed) ed[(long)(rbase + i) * H + h] = sd;
        }
      }
    }
  }
}

// ----- edge kernels --------------------------------------------------------

__global__ void fill_f32(float* __restrict__ p, float v, long n) {
  long i = (long)blockIdx.x * blockDim.x + threadIdx.x;
  if (i < n) p[i] = v;
}

template<int H>
__global__ void edge_logits_k(const int* __restrict__ src, const int* __restrict__ dst,
                              const float* __restrict__ es, const float* __restrict__ ed,
                              float* __restrict__ elog, float* __restrict__ m, long E)
{
  long e = (long)blockIdx.x * blockDim.x + threadIdx.x;
  if (e >= E) return;
  int s = src[e], d = dst[e];
  #pragma unroll
  for (int h = 0; h < H; ++h) {
    float v = es[(long)s * H + h] + ed[(long)d * H + h];
    v = v > 0.f ? v : 0.2f * v;           // leaky_relu(0.2)
    elog[e * H + h] = v;
    atomicMax(&m[(long)d * H + h], v);    // segment max
  }
}

template<int H>
__global__ void edge_exp_k(const int* __restrict__ dst, const float* __restrict__ m,
                           float* __restrict__ elog, float* __restrict__ den, long E)
{
  long e = (long)blockIdx.x * blockDim.x + threadIdx.x;
  if (e >= E) return;
  int d = dst[e];
  #pragma unroll
  for (int h = 0; h < H; ++h) {
    float ex = expf(elog[e * H + h] - m[(long)d * H + h]);
    elog[e * H + h] = ex;
    atomicAdd(&den[(long)d * H + h], ex); // segment sum
  }
}

template<int H, int C>
__global__ void edge_aggr_k(const int* __restrict__ src, const int* __restrict__ dst,
                            const float* __restrict__ hs, const float* __restrict__ ex,
                            const float* __restrict__ den, float* __restrict__ out, long E)
{
  long idx = (long)blockIdx.x * blockDim.x + threadIdx.x;
  if (idx >= E * (long)H) return;
  long e = idx / H;
  int  h = (int)(idx - e * H);
  int s = src[e], d = dst[e];
  float alpha = ex[e * H + h] / (den[(long)d * H + h] + 1e-16f);
  const float* hp = hs + (long)s * (H * C) + h * C;
  float*       op = out + (long)d * (H * C) + h * C;
  #pragma unroll
  for (int c = 0; c < C; ++c) atomicAdd(&op[c], alpha * hp[c]);
}

// ----- elementwise epilogues ----------------------------------------------

// out = elu((A + bA + B + bB) * 0.5) (+ res)
__global__ void combine_mean_elu_res(const float* __restrict__ A, const float* __restrict__ bA,
                                     const float* __restrict__ B, const float* __restrict__ bB,
                                     const float* __restrict__ res, float* __restrict__ out,
                                     int F, long n)
{
  long i = (long)blockIdx.x * blockDim.x + threadIdx.x;
  if (i >= n) return;
  int c = (int)(i % F);
  float v = eluf((A[i] + bA[c] + B[i] + bB[c]) * 0.5f);
  out[i] = res ? v + res[i] : v;
}

__global__ void bias_elu(const float* __restrict__ A, const float* __restrict__ b,
                         float* __restrict__ out, int F, long n)
{
  long i = (long)blockIdx.x * blockDim.x + threadIdx.x;
  if (i >= n) return;
  int c = (int)(i % F);
  out[i] = eluf(A[i] + b[c]);
}

__global__ void build_cat36(const float* __restrict__ ue, const float* __restrict__ qol,
                            float* __restrict__ cat, long M)
{
  long i = (long)blockIdx.x * blockDim.x + threadIdx.x;
  if (i >= M * 36) return;
  long r = i / 36;
  int  c = (int)(i - r * 36);
  cat[i] = (c < 32) ? ue[r * 32 + c] : qol[r * 4 + (c - 32)];
}

__global__ void final_user_k(const float* __restrict__ ue, const float* __restrict__ gate_raw,
                             const float* __restrict__ qsig, float* __restrict__ out, long n)
{
  long i = (long)blockIdx.x * blockDim.x + threadIdx.x;
  if (i >= n) return;
  float g = 1.f / (1.f + expf(-gate_raw[i]));
  out[i] = ue[i] + 0.3f * g * qsig[i];
}

// ----- host orchestration --------------------------------------------------

static inline int cdiv(long a, int b) { return (int)((a + b - 1) / b); }

template<int NT, int K>
static void launch_pack(const float* W, __bf16* P, hipStream_t st) {
  constexpr int KT = (K + 31) & ~31;
  constexpr int TOTAL = (KT / 32) * NT * 512;
  pack_w_k<NT, K><<<(TOTAL + 255) / 256, 256, 0, st>>>(W, P);
}

template<int NT, int H, int MODE, int K>
static void launch_gemm(const float* X, const __bf16* Wp, const float* bias,
                        const float* g, const float* be,
                        const float* as_, const float* ad_,
                        float* Y, float* es, float* ed, int M, hipStream_t st)
{
  int blocks = ((M + 15) / 16 + 7) / 8;   // 8 waves / block, 1 stripe / wave
  gemm_k<NT, H, MODE, K><<<blocks, 256, 0, st>>>(X, Wp, bias, g, be, as_, ad_, Y, es, ed, M);
}

template<int H, int C>
static void run_edge_softmax_aggr(const int* src, const int* dst, long E, int Nd,
                                  const float* hs, const float* es, const float* ed,
                                  float* elog, float* m, float* den, float* out, hipStream_t st)
{
  long mh = (long)Nd * H;
  long oc = (long)Nd * H * C;
  fill_f32<<<cdiv(mh, 256), 256, 0, st>>>(m, -INFINITY, mh);
  fill_f32<<<cdiv(mh, 256), 256, 0, st>>>(den, 0.f, mh);
  fill_f32<<<cdiv(oc, 256), 256, 0, st>>>(out, 0.f, oc);
  edge_logits_k<H><<<cdiv(E, 256), 256, 0, st>>>(src, dst, es, ed, elog, m, E);
  edge_exp_k<H><<<cdiv(E, 256), 256, 0, st>>>(dst, m, elog, den, E);
  edge_aggr_k<H, C><<<cdiv(E * H, 256), 256, 0, st>>>(src, dst, hs, elog, den, out, E);
}

extern "C" void kernel_launch(void* const* d_in, const int* in_sizes, int n_in,
                              void* d_out, int out_size, void* d_ws, size_t ws_size,
                              hipStream_t stream)
{
  (void)in_sizes; (void)n_in; (void)out_size; (void)ws_size;
  #define FIN(i) ((const float*)d_in[i])
  const float* x_user = FIN(0);
  const float* x_item = FIN(1);
  const float* qol    = FIN(2);
  const int*   e_soc  = (const int*)d_in[3];
  const int*   e_int  = (const int*)d_in[4];
  const float *up_w = FIN(5),  *up_b = FIN(6),  *up_g = FIN(7),  *up_beta = FIN(8);
  const float *ip_w1 = FIN(9), *ip_b1 = FIN(10), *ip_g = FIN(11), *ip_beta = FIN(12);
  const float *ip_w2 = FIN(13), *ip_b2 = FIN(14);
  const float *soc1_w = FIN(15), *soc1_as = FIN(16), *soc1_ad = FIN(17), *soc1_b = FIN(18);
  const float *int1_ws = FIN(19), *int1_wd = FIN(20), *int1_as = FIN(21), *int1_ad = FIN(22), *int1_b = FIN(23);
  const float *rev1_ws = FIN(24), *rev1_wd = FIN(25), *rev1_as = FIN(26), *rev1_ad = FIN(27), *rev1_b = FIN(28);
  const float *soc2_w = FIN(29), *soc2_as = FIN(30), *soc2_ad = FIN(31), *soc2_b = FIN(32);
  const float *int2_ws = FIN(33), *int2_wd = FIN(34), *int2_as = FIN(35), *int2_ad = FIN(36), *int2_b = FIN(37);
  const float *rev2_ws = FIN(38), *rev2_wd = FIN(39), *rev2_as = FIN(40), *rev2_ad = FIN(41), *rev2_b = FIN(42);
  const float *qg_w = FIN(43), *qg_b = FIN(44), *qp_w = FIN(45), *qp_b = FIN(46);
  #undef FIN

  const int* ss = e_soc;             // social src
  const int* sd = e_soc + ES_N;      // social dst
  const int* iu = e_int;             // interact user (src)
  const int* ii = e_int + EI_N;      // interact item (dst)

  // --- workspace layout (floats) ---
  float* ws   = (float*)d_ws;
  float* hu   = ws;                           // NU*64
  float* hi   = hu   + (long)NU_ * 64;        // NI*64
  float* h1u  = hi   + (long)NI_ * 64;        // NU*64
  float* h1i  = h1u  + (long)NU_ * 64;        // NI*64
  float* ue2  = h1i  + (long)NI_ * 64;        // NU*32
  float* HS   = ue2  + (long)NU_ * 32;        // NU*64 (src projections / t128 / cat36)
  float* HD   = HS   + (long)NU_ * 64;        // NU*64 (dst projections / gate)
  float* ESb  = HD   + (long)NU_ * 64;        // NU*4
  float* EDb  = ESb  + (long)NU_ * 4;         // NU*4
  float* MM   = EDb  + (long)NU_ * 4;         // NU*4
  float* DEN  = MM   + (long)NU_ * 4;         // NU*4
  float* ELOG = DEN  + (long)NU_ * 4;         // EI*4 (edge logits/exp / qsig)
  float* OUTA = ELOG + (long)EI_N * 4;        // NU*64
  float* OUTB = OUTA + (long)NU_ * 64;        // NU*64
  float* T128 = HS;    // NI*128 == 6.4e6 floats, fits the HS slot (used before convs)
  float* CAT  = HS;    // NU*36, after convs
  float* GATE = HD;    // NU*32, after convs
  float* QSIG = ELOG;  // NU*32, after convs

  // packed bf16 weights after OUTB (offset is a multiple of 32B)
  __bf16* PK     = (__bf16*)(OUTB + (long)NU_ * 64);
  __bf16* P_UP   = PK;               // KT=64 * 64  = 4096
  __bf16* P_IP1  = P_UP   + 4096;    // 384*128     = 49152
  __bf16* P_IP2  = P_IP1  + 49152;   // 128*64      = 8192
  __bf16* P_SOC1 = P_IP2  + 8192;    // 64*64       = 4096
  __bf16* P_INT1S= P_SOC1 + 4096;
  __bf16* P_INT1D= P_INT1S+ 4096;
  __bf16* P_REV1S= P_INT1D+ 4096;
  __bf16* P_REV1D= P_REV1S+ 4096;
  __bf16* P_SOC2 = P_REV1D+ 4096;    // 64*32       = 2048
  __bf16* P_INT2S= P_SOC2 + 2048;
  __bf16* P_INT2D= P_INT2S+ 2048;
  __bf16* P_REV2S= P_INT2D+ 2048;
  __bf16* P_REV2D= P_REV2S+ 2048;
  __bf16* P_QG   = P_REV2D+ 2048;    // KT=64 * 32  = 2048
  __bf16* P_QP   = P_QG   + 2048;    // KT=32 * 32  = 1024

  // --- pack all weights into WMMA B-fragment order (bf16) ---
  launch_pack<4, 36 >(up_w,    P_UP,   stream);
  launch_pack<8, 384>(ip_w1,   P_IP1,  stream);
  launch_pack<4, 128>(ip_w2,   P_IP2,  stream);
  launch_pack<4, 64 >(soc1_w,  P_SOC1, stream);
  launch_pack<4, 64 >(int1_ws, P_INT1S,stream);
  launch_pack<4, 64 >(int1_wd, P_INT1D,stream);
  launch_pack<4, 64 >(rev1_ws, P_REV1S,stream);
  launch_pack<4, 64 >(rev1_wd, P_REV1D,stream);
  launch_pack<2, 64 >(soc2_w,  P_SOC2, stream);
  launch_pack<2, 64 >(int2_ws, P_INT2S,stream);
  launch_pack<2, 64 >(int2_wd, P_INT2D,stream);
  launch_pack<2, 64 >(rev2_ws, P_REV2S,stream);
  launch_pack<2, 64 >(rev2_wd, P_REV2D,stream);
  launch_pack<2, 36 >(qg_w,    P_QG,   stream);
  launch_pack<2, 4  >(qp_w,    P_QP,   stream);

  // --- input projections ---
  // hu = gelu(LN(x_user@up_w + up_b))        [NU,64]
  launch_gemm<4, 1, 1, 36>(x_user, P_UP, up_b, up_g, up_beta, nullptr, nullptr,
                           hu, nullptr, nullptr, NU_, stream);
  // t = gelu(LN(x_item@ip_w1 + ip_b1))       [NI,128]
  launch_gemm<8, 1, 1, 384>(x_item, P_IP1, ip_b1, ip_g, ip_beta, nullptr, nullptr,
                            T128, nullptr, nullptr, NI_, stream);
  // hi = t@ip_w2 + ip_b2                     [NI,64]
  launch_gemm<4, 1, 0, 128>(T128, P_IP2, ip_b2, nullptr, nullptr, nullptr, nullptr,
                            hi, nullptr, nullptr, NI_, stream);

  // --- layer 1 (H=4, C=16) ---
  // soc1: u->u (shared projection)
  launch_gemm<4, 4, 2, 64>(hu, P_SOC1, nullptr, nullptr, nullptr, soc1_as, soc1_ad,
                           HS, ESb, EDb, NU_, stream);
  run_edge_softmax_aggr<4, 16>(ss, sd, ES_N, NU_, HS, ESb, EDb, ELOG, MM, DEN, OUTA, stream);
  // rev1: item->user
  launch_gemm<4, 4, 2, 64>(hi, P_REV1S, nullptr, nullptr, nullptr, rev1_as, nullptr,
                           HS, ESb, nullptr, NI_, stream);
  launch_gemm<4, 4, 2, 64>(hu, P_REV1D, nullptr, nullptr, nullptr, nullptr, rev1_ad,
                           HD, nullptr, EDb, NU_, stream);
  run_edge_softmax_aggr<4, 16>(ii, iu, EI_N, NU_, HS, ESb, EDb, ELOG, MM, DEN, OUTB, stream);
  // h1u = elu((u_soc + u_rev)*0.5) + hu
  {
    long n = (long)NU_ * 64;
    combine_mean_elu_res<<<cdiv(n, 256), 256, 0, stream>>>(OUTA, soc1_b, OUTB, rev1_b, hu, h1u, 64, n);
  }
  // int1: user->item
  launch_gemm<4, 4, 2, 64>(hu, P_INT1S, nullptr, nullptr, nullptr, int1_as, nullptr,
                           HS, ESb, nullptr, NU_, stream);
  launch_gemm<4, 4, 2, 64>(hi, P_INT1D, nullptr, nullptr, nullptr, nullptr, int1_ad,
                           HD, nullptr, EDb, NI_, stream);
  run_edge_softmax_aggr<4, 16>(iu, ii, EI_N, NI_, HS, ESb, EDb, ELOG, MM, DEN, OUTA, stream);
  {
    long n = (long)NI_ * 64;
    bias_elu<<<cdiv(n, 256), 256, 0, stream>>>(OUTA, int1_b, h1i, 64, n);
  }

  // --- layer 2 (H=1, C=32) ---
  // soc2: u->u (shared projection)
  launch_gemm<2, 1, 2, 64>(h1u, P_SOC2, nullptr, nullptr, nullptr, soc2_as, soc2_ad,
                           HS, ESb, EDb, NU_, stream);
  run_edge_softmax_aggr<1, 32>(ss, sd, ES_N, NU_, HS, ESb, EDb, ELOG, MM, DEN, OUTA, stream);
  // rev2: item->user
  launch_gemm<2, 1, 2, 64>(h1i, P_REV2S, nullptr, nullptr, nullptr, rev2_as, nullptr,
                           HS, ESb, nullptr, NI_, stream);
  launch_gemm<2, 1, 2, 64>(h1u, P_REV2D, nullptr, nullptr, nullptr, nullptr, rev2_ad,
                           HD, nullptr, EDb, NU_, stream);
  run_edge_softmax_aggr<1, 32>(ii, iu, EI_N, NU_, HS, ESb, EDb, ELOG, MM, DEN, OUTB, stream);
  {
    long n = (long)NU_ * 32;
    combine_mean_elu_res<<<cdiv(n, 256), 256, 0, stream>>>(OUTA, soc2_b, OUTB, rev2_b, nullptr, ue2, 32, n);
  }
  // int2: user->item -> item_emb directly into d_out
  launch_gemm<2, 1, 2, 64>(h1u, P_INT2S, nullptr, nullptr, nullptr, int2_as, nullptr,
                           HS, ESb, nullptr, NU_, stream);
  launch_gemm<2, 1, 2, 64>(h1i, P_INT2D, nullptr, nullptr, nullptr, nullptr, int2_ad,
                           HD, nullptr, EDb, NI_, stream);
  run_edge_softmax_aggr<1, 32>(iu, ii, EI_N, NI_, HS, ESb, EDb, ELOG, MM, DEN, OUTA, stream);
  float* out_user = (float*)d_out;
  float* out_item = out_user + (long)NU_ * 32;
  {
    long n = (long)NI_ * 32;
    bias_elu<<<cdiv(n, 256), 256, 0, stream>>>(OUTA, int2_b, out_item, 32, n);
  }

  // --- QoL gating ---
  launch_gemm<2, 1, 0, 4>(qol, P_QP, qp_b, nullptr, nullptr, nullptr, nullptr,
                          QSIG, nullptr, nullptr, NU_, stream);
  {
    long n = (long)NU_ * 36;
    build_cat36<<<cdiv(n, 256), 256, 0, stream>>>(ue2, qol, CAT, NU_);
  }
  launch_gemm<2, 1, 0, 36>(CAT, P_QG, qg_b, nullptr, nullptr, nullptr, nullptr,
                           GATE, nullptr, nullptr, NU_, stream);
  {
    long n = (long)NU_ * 32;
    final_user_k<<<cdiv(n, 256), 256, 0, stream>>>(ue2, GATE, QSIG, out_user, n);
  }
}
